// GINClassifier_1769526526272
// MI455X (gfx1250) — compile-verified, matrix operand order checked by reading, not run on hardware
//
#include <hip/hip_runtime.h>
#include <hip/hip_bf16.h>

#define N_NODES   50000
#define N_EDGES   800000
#define D_H       128
#define N_LAYERS  3
#define N_CLASSES 10
#define NUM_GRAPHS 256
#define BN_EPS    1e-5f

// padded LDS row stride (dwords): 68 = 64+4 -> bank stride 4, conflict-free
// b128 reads across a 16-lane half; 68 % 4 == 0 keeps uint4 alignment.
#define LSTRIDE 68

typedef __attribute__((ext_vector_type(16))) __bf16 v16bf;
typedef __attribute__((ext_vector_type(8)))  float  v8f;

union BfPack {
    unsigned int u[8];
    v16bf v;
};

__device__ __forceinline__ unsigned int f2bf_rne(float f) {
    unsigned int u = __float_as_uint(f);
    unsigned int r = u + 0x7FFFu + ((u >> 16) & 1u);
    return r >> 16;
}
__device__ __forceinline__ unsigned int pack2bf(float lo, float hi) {
    return (f2bf_rne(hi) << 16) | f2bf_rne(lo);
}

// ---------------------------------------------------------------- zero fill
__global__ void k_zero(float* __restrict__ p, long n) {
    long i = (long)blockIdx.x * blockDim.x + threadIdx.x;
    long stride = (long)gridDim.x * blockDim.x;
    for (; i < n; i += stride) p[i] = 0.0f;
}

// ------------------------------------------------------ edge scatter (GIN agg)
// one 32-lane wave per edge; each lane moves 4 channels with float4 gather
__global__ void k_scatter(const float* __restrict__ H,
                          const int* __restrict__ src,
                          const int* __restrict__ dst,
                          float* __restrict__ AGG) {
    long g = (long)blockIdx.x * blockDim.x + threadIdx.x;
    int edge = (int)(g >> 5);
    int lane = (int)(g & 31);
    if (edge >= N_EDGES) return;
    int s = src[edge];
    int d = dst[edge];
    const float4 v = ((const float4*)(H + (long)s * D_H))[lane];
    float* ap = AGG + (long)d * D_H + lane * 4;
    atomicAdd(ap + 0, v.x);
    atomicAdd(ap + 1, v.y);
    atomicAdd(ap + 2, v.z);
    atomicAdd(ap + 3, v.w);
}

// ----------------------------------------------- WMMA GEMM + fused BN stats
// Z[M,128] = (X [+X2]) @ W[128,128] + bias ; colsum/colsumsq accumulated.
// block = 128 threads = 4 waves, each wave owns a 16x128 strip (64 rows/block)
__global__ void k_gemm_bnstats(const float* __restrict__ X,
                               const float* __restrict__ X2, // optional add (agg), may be null
                               const float* __restrict__ W,
                               const float* __restrict__ bias,
                               float* __restrict__ Z,
                               float* __restrict__ colsum,
                               float* __restrict__ colsumsq,
                               int M) {
    // W transposed+packed: wpT[col*LSTRIDE + q] = bf16pair(W[2q][col], W[2q+1][col])
    __shared__ unsigned int wpT[128 * LSTRIDE];
    // X packed: xp[row*LSTRIDE + q] = bf16pair(X[row][2q], X[row][2q+1])
    __shared__ unsigned int xp[64 * LSTRIDE];

    const int tid = threadIdx.x;
    const int blockRow = blockIdx.x * 64;

    // stage weights (packed bf16 pairs along K, transposed for contiguous B reads)
    for (int e = tid; e < 8192; e += 128) {
        int q = e >> 7, col = e & 127;
        float w0 = W[(2 * q) * 128 + col];
        float w1 = W[(2 * q + 1) * 128 + col];
        wpT[col * LSTRIDE + q] = pack2bf(w0, w1);
    }
    // stage activations (with optional fused h+agg), zero-pad tail rows
    for (int e = tid; e < 4096; e += 128) {
        int rowL = e >> 6, q = e & 63;
        int row = blockRow + rowL;
        unsigned int pk = 0u;
        if (row < M) {
            long off = (long)row * 128 + 2 * q;
            float a0 = X[off], a1 = X[off + 1];
            if (X2) { a0 += X2[off]; a1 += X2[off + 1]; }
            pk = pack2bf(a0, a1);
        }
        xp[rowL * LSTRIDE + q] = pk;
    }
    __syncthreads();

    const int wave = tid >> 5;
    const int lane = tid & 31;
    const int ln   = lane & 15;
    const int half = lane >> 4;
    const int ph   = half * 4;
    const int rowL = wave * 16 + ln;

    v8f acc[8];
#pragma unroll
    for (int nt = 0; nt < 8; ++nt) acc[nt] = (v8f){0, 0, 0, 0, 0, 0, 0, 0};

#pragma unroll
    for (int kk = 0; kk < 4; ++kk) {
        const int abase = rowL * LSTRIDE + kk * 16 + ph;
        BfPack a;
        {   // two contiguous b128 reads, bank-conflict-free (stride 68)
            uint4 a0 = *(const uint4*)&xp[abase];
            uint4 a1 = *(const uint4*)&xp[abase + 8];
            a.u[0] = a0.x; a.u[1] = a0.y; a.u[2] = a0.z; a.u[3] = a0.w;
            a.u[4] = a1.x; a.u[5] = a1.y; a.u[6] = a1.z; a.u[7] = a1.w;
        }
#pragma unroll
        for (int nt = 0; nt < 8; ++nt) {
            const int cbase = (nt * 16 + ln) * LSTRIDE + kk * 16 + ph;
            BfPack b;
            uint4 b0 = *(const uint4*)&wpT[cbase];
            uint4 b1 = *(const uint4*)&wpT[cbase + 8];
            b.u[0] = b0.x; b.u[1] = b0.y; b.u[2] = b0.z; b.u[3] = b0.w;
            b.u[4] = b1.x; b.u[5] = b1.y; b.u[6] = b1.z; b.u[7] = b1.w;
            acc[nt] = __builtin_amdgcn_wmma_f32_16x16x32_bf16(
                false, a.v, false, b.v, (short)0, acc[nt], false, false);
        }
    }

    // epilogue: bias, store, per-column partial stats (one atomic pair / tile)
    const int gr0 = blockRow + wave * 16 + half * 8;
#pragma unroll
    for (int nt = 0; nt < 8; ++nt) {
        int col = nt * 16 + ln;
        float bcol = bias[col];
        float s = 0.0f, s2 = 0.0f;
#pragma unroll
        for (int r = 0; r < 8; ++r) {
            int row = gr0 + r;
            if (row < M) {
                float val = acc[nt][r] + bcol;
                Z[(long)row * 128 + col] = val;
                s += val;
                s2 += val * val;
            }
        }
        atomicAdd(&colsum[col], s);
        atomicAdd(&colsumsq[col], s2);
    }
}

// ------------------------------------------------------------- BN + ReLU apply
__global__ void k_bn_relu(const float* __restrict__ Zin,
                          float* __restrict__ Zout,
                          const float* __restrict__ colsum,
                          const float* __restrict__ colsumsq,
                          const float* __restrict__ gamma,
                          const float* __restrict__ beta) {
    long i = (long)blockIdx.x * blockDim.x + threadIdx.x;
    long n = (long)N_NODES * D_H;
    long stride = (long)gridDim.x * blockDim.x;
    const float invN = 1.0f / (float)N_NODES;
    for (; i < n; i += stride) {
        int c = (int)(i & 127);
        float mu  = colsum[c] * invN;
        float var = colsumsq[c] * invN - mu * mu;
        float z = (Zin[i] - mu) * rsqrtf(var + BN_EPS) * gamma[c] + beta[c];
        Zout[i] = z > 0.0f ? z : 0.0f;
    }
}

// ------------------------------------------------------------- graph readout
__global__ void k_readout(const float* __restrict__ H,
                          const int* __restrict__ batch,
                          float* __restrict__ G, int layer) {
    long g = (long)blockIdx.x * blockDim.x + threadIdx.x;
    int node = (int)(g >> 5);
    int lane = (int)(g & 31);
    if (node >= N_NODES) return;
    int b = batch[node];
    const float4 v = ((const float4*)(H + (long)node * D_H))[lane];
    float* gp = G + (long)b * (N_LAYERS * D_H) + layer * D_H + lane * 4;
    atomicAdd(gp + 0, v.x);
    atomicAdd(gp + 1, v.y);
    atomicAdd(gp + 2, v.z);
    atomicAdd(gp + 3, v.w);
}

// ------------------------------------------------------------- tiny classifier
__global__ void k_classifier(const float* __restrict__ G,
                             const float* __restrict__ Wc1,
                             const float* __restrict__ bc1,
                             const float* __restrict__ Wc2,
                             const float* __restrict__ bc2,
                             float* __restrict__ out) {
    __shared__ float hg[N_LAYERS * D_H];
    __shared__ float y[D_H];
    const int g = blockIdx.x;
    const int t = threadIdx.x; // 128 threads
    for (int k = t; k < N_LAYERS * D_H; k += 128)
        hg[k] = G[(long)g * (N_LAYERS * D_H) + k];
    __syncthreads();
    float a = bc1[t];
    for (int k = 0; k < N_LAYERS * D_H; ++k)
        a += hg[k] * Wc1[k * D_H + t];
    y[t] = a > 0.0f ? a : 0.0f;
    __syncthreads();
    if (t < N_CLASSES) {
        float o = bc2[t];
        for (int c = 0; c < D_H; ++c)
            o += y[c] * Wc2[c * N_CLASSES + t];
        out[g * N_CLASSES + t] = o;
    }
}

extern "C" void kernel_launch(void* const* d_in, const int* in_sizes, int n_in,
                              void* d_out, int out_size, void* d_ws, size_t ws_size,
                              hipStream_t stream) {
    const float* x    = (const float*)d_in[0];
    const int*   edge = (const int*)d_in[1];      // [2, E]: row0 = src, row1 = dst
    const int*   bat  = (const int*)d_in[2];
    const float* W1   = (const float*)d_in[3];
    const float* b1   = (const float*)d_in[4];
    const float* g1   = (const float*)d_in[5];
    const float* bt1  = (const float*)d_in[6];
    const float* W2   = (const float*)d_in[7];
    const float* b2   = (const float*)d_in[8];
    const float* g2   = (const float*)d_in[9];
    const float* bt2  = (const float*)d_in[10];
    const float* Wc1  = (const float*)d_in[11];
    const float* bc1  = (const float*)d_in[12];
    const float* Wc2  = (const float*)d_in[13];
    const float* bc2  = (const float*)d_in[14];
    float* out = (float*)d_out;

    const long NF = (long)N_NODES * D_H;
    float* ws  = (float*)d_ws;
    float* AGG = ws;                // [N,128] agg, reused as Z2
    float* Z   = ws + NF;           // [N,128]
    float* H   = ws + 2 * NF;       // [N,128] layer output features
    float* S   = ws + 3 * NF;       // 512 floats: sum1,sq1,sum2,sq2
    float* G   = S + 512;           // [256, 384] readout concat

    const int* src = edge;
    const int* dst = edge + N_EDGES;

    dim3 blk(256);
    const int zeroGridNF = 4096;
    const int scatGrid   = (N_EDGES * 32 + 255) / 256;
    const int gemmGrid   = (N_NODES + 63) / 64;
    const int bnGrid     = 8192;
    const int roGrid     = (N_NODES * 32 + 255) / 256;

    k_zero<<<48, blk, 0, stream>>>(G, (long)NUM_GRAPHS * N_LAYERS * D_H);

    const float* Hprev = x;
    for (int i = 0; i < N_LAYERS; ++i) {
        k_zero<<<zeroGridNF, blk, 0, stream>>>(AGG, NF);
        k_zero<<<2, blk, 0, stream>>>(S, 512);
        k_scatter<<<scatGrid, blk, 0, stream>>>(Hprev, src, dst, AGG);
        // z = (h + agg) @ W1 + b1, stats into S[0:256]
        k_gemm_bnstats<<<gemmGrid, dim3(128), 0, stream>>>(
            Hprev, AGG, W1 + (long)i * D_H * D_H, b1 + i * D_H,
            Z, S, S + 128, N_NODES);
        k_bn_relu<<<bnGrid, blk, 0, stream>>>(Z, Z, S, S + 128,
                                              g1 + i * D_H, bt1 + i * D_H);
        // z2 = z @ W2 + b2 (into AGG), stats into S[256:512]
        k_gemm_bnstats<<<gemmGrid, dim3(128), 0, stream>>>(
            Z, nullptr, W2 + (long)i * D_H * D_H, b2 + i * D_H,
            AGG, S + 256, S + 384, N_NODES);
        k_bn_relu<<<bnGrid, blk, 0, stream>>>(AGG, H, S + 256, S + 384,
                                              g2 + i * D_H, bt2 + i * D_H);
        k_readout<<<roGrid, blk, 0, stream>>>(H, bat, G, i);
        Hprev = H;
    }

    k_classifier<<<NUM_GRAPHS, dim3(128), 0, stream>>>(G, Wc1, bc1, Wc2, bc2, out);
}
